// VisionCore_54975581389314
// MI455X (gfx1250) — compile-verified, matrix operand order checked by reading, not run on hardware
//
#include <hip/hip_runtime.h>
#include <hip/hip_bf16.h>

// ---------------------------------------------------------------------------
// Types for WMMA operands
// ---------------------------------------------------------------------------
typedef _Float16 v16h __attribute__((ext_vector_type(16)));
typedef _Float16 v8h  __attribute__((ext_vector_type(8)));
typedef float    v8f  __attribute__((ext_vector_type(8)));

// f16 transposed-weight layout inside workspace (element offsets in halfs)
#define OFF0 0          // Wt0 [256][256]
#define OFF1 65536      // Wt1 [128][256]
#define OFF2 98304      // Wt2 [ 64][128]
#define OFF3 106496     // Wt3 [ 32][ 64]
#define OFF4 108544     // Wt4 [ 32][ 32]
#define WT_HALFS 109568
#define WT_BYTES (WT_HALFS * 2)   // 219136, multiple of 256

// ---------------------------------------------------------------------------
// Kernel 1: transpose f32 weight (K x N) -> f16 (N x K)
// ---------------------------------------------------------------------------
__global__ void wprep_kernel(const float* __restrict__ W, _Float16* __restrict__ Wt,
                             int Kd, int Nd) {
    int i = blockIdx.x * 256 + threadIdx.x;
    if (i < Kd * Nd) {
        int k = i / Nd, n = i % Nd;
        Wt[n * Kd + k] = (_Float16)W[i];
    }
}

// ---------------------------------------------------------------------------
// One MLP layer on a 64-row batch tile via v_wmma_f32_16x16x32_f16.
// A fragment (16x32 f16): lane l holds row (l&15); K chunks [base, base+8) and
// [base+16, base+24) with base = kstep*32 + (l>>4)*8  (ISA 7.12.2 layout).
// B fragment (32x16 f16): same pattern with lane -> output column, sourced
// from transposed weights Wt[N][K] so both chunk loads are contiguous 16B.
// C/D (16x16 f32, 8 VGPRs): element r of acc is row r + (l>>4)*8, col l&15.
// ---------------------------------------------------------------------------
template <int K, int N, bool RELU, bool TO_LDS>
__device__ inline void mlp_layer(const _Float16* __restrict__ wt,   // global [N][K]
                                 const float* __restrict__ bias,    // global [N]
                                 const _Float16* in_lds,            // LDS [64][K]
                                 _Float16* out_lds,                 // LDS [64][N]
                                 float* __restrict__ qout,          // global (final)
                                 int rowBase, int wave, int lane) {
    constexpr int KS = K / 32;
    constexpr int NT = N / 16;
    const int lmod = lane & 15;
    const int lhi  = lane >> 4;     // 0 or 1
    const int kOff = lhi * 8;

    for (int nt = wave; nt < NT; nt += 8) {
        const int ncol = nt * 16 + lmod;
        // Cache all K-step B fragments for this N-tile in VGPRs (reused 4x).
        v16h bfrag[KS];
#pragma unroll
        for (int ks = 0; ks < KS; ++ks) {
            const _Float16* p = wt + (size_t)ncol * K + ks * 32 + kOff;
            union { v16h v; v8h h[2]; } u;
            u.h[0] = *(const v8h*)(p);
            u.h[1] = *(const v8h*)(p + 16);
            bfrag[ks] = u.v;
        }
        const float bs = bias[ncol];

#pragma unroll
        for (int mt = 0; mt < 4; ++mt) {
            v8f acc = {};
#pragma unroll
            for (int ks = 0; ks < KS; ++ks) {
                const _Float16* p = in_lds + (mt * 16 + lmod) * K + ks * 32 + kOff;
                union { v16h v; v8h h[2]; } u;
                u.h[0] = *(const v8h*)(p);       // ds_load_b128
                u.h[1] = *(const v8h*)(p + 16);  // ds_load_b128
                acc = __builtin_amdgcn_wmma_f32_16x16x32_f16(
                    false, u.v, false, bfrag[ks], (short)0, acc, false, false);
            }
#pragma unroll
            for (int r = 0; r < 8; ++r) {
                float v = acc[r] + bs;
                if (RELU) v = v > 0.f ? v : 0.f;
                const int m = mt * 16 + r + lhi * 8;
                if (TO_LDS) out_lds[m * N + ncol] = (_Float16)v;
                else        qout[(size_t)(rowBase + m) * N + ncol] = v;
            }
        }
    }
}

// ---------------------------------------------------------------------------
// Kernel 2: q-MLP. 64 batch rows / workgroup, 8 waves, LDS ping-pong (64 KB).
// ---------------------------------------------------------------------------
__global__ void __launch_bounds__(256)
mlp_kernel(const float* __restrict__ hidden, const _Float16* __restrict__ wt,
           const float* __restrict__ b0, const float* __restrict__ b1,
           const float* __restrict__ b2, const float* __restrict__ b3,
           const float* __restrict__ b4, float* __restrict__ qout) {
    __shared__ alignas(16) _Float16 bufA[64 * 256];
    __shared__ alignas(16) _Float16 bufB[64 * 256];

    const int tid  = threadIdx.x;
    const int wave = tid >> 5;
    const int lane = tid & 31;
    const int rowBase = blockIdx.x * 64;

    // Stage hidden state (f32 -> f16) into LDS, row-major [64][256].
    const float* src = hidden + (size_t)rowBase * 256;
    for (int i = tid; i < 64 * 256; i += 256) bufA[i] = (_Float16)src[i];
    __syncthreads();

    mlp_layer<256, 256, true,  true >(wt + OFF0, b0, bufA, bufB, qout, rowBase, wave, lane);
    __syncthreads();
    mlp_layer<256, 128, true,  true >(wt + OFF1, b1, bufB, bufA, qout, rowBase, wave, lane);
    __syncthreads();
    mlp_layer<128,  64, true,  true >(wt + OFF2, b2, bufA, bufB, qout, rowBase, wave, lane);
    __syncthreads();
    mlp_layer< 64,  32, true,  true >(wt + OFF3, b3, bufB, bufA, qout, rowBase, wave, lane);
    __syncthreads();
    mlp_layer< 32,  32, false, false>(wt + OFF4, b4, bufA, bufB, qout, rowBase, wave, lane);
}

// ---------------------------------------------------------------------------
// Kernel 3: convs + attention + head. 16 samples x 16 threads per workgroup.
// ---------------------------------------------------------------------------
__global__ void __launch_bounds__(256)
conv_attn_kernel(const float* __restrict__ X,
                 const float* __restrict__ c1w, const float* __restrict__ c1b,
                 const float* __restrict__ c2w, const float* __restrict__ c2b,
                 const float* __restrict__ qws,
                 const float* __restrict__ aw0, const float* __restrict__ ab0,
                 const float* __restrict__ aw1, const float* __restrict__ ab1,
                 float* __restrict__ out) {
    constexpr int SPW = 16;
    __shared__ float Xs[SPW * 147];   // 7x7x3 per sample
    __shared__ float O1[SPW * 200];   // 5x5x8
    __shared__ float O2[SPW * 300];   // 5x5x12
    __shared__ float Sb[100];         // 5x5x4 spatial basis
    __shared__ float W1[600];
    __shared__ float Bi1[8];
    __shared__ float W2[864];
    __shared__ float Bi2[12];
    __shared__ float Aq[SPW * 48];
    __shared__ float Hh[SPW * 32];

    const int tid = threadIdx.x;
    const int s = tid >> 4, t = tid & 15;
    const size_t bbase = (size_t)blockIdx.x * SPW;
    const int b = (int)bbase + s;

    for (int i = tid; i < SPW * 147; i += 256) Xs[i] = X[bbase * 147 + i];
    for (int i = tid; i < 600; i += 256) W1[i] = c1w[i];
    for (int i = tid; i < 864; i += 256) W2[i] = c2w[i];
    if (tid < 8)  Bi1[tid] = c1b[tid];
    if (tid < 12) Bi2[tid] = c2b[tid];
    if (tid < 100) {
        const int p = tid >> 2, uv = tid & 3;
        const int h = p / 5, w = p % 5;
        const int u = uv >> 1, v = uv & 1;
        const float PI5 = 0.628318530717958648f;  // pi/5
        Sb[tid] = __cosf((float)(h + 1) * PI5 * (float)(u + 1)) *
                  __cosf((float)(w + 1) * PI5 * (float)(v + 1));
    }
    __syncthreads();

    // conv1: 5x5x3 -> 8ch, pad 1 (7x7 -> 5x5)
    for (int p = t; p < 25; p += 16) {
        const int oy = p / 5, ox = p % 5;
        float acc[8];
#pragma unroll
        for (int c = 0; c < 8; ++c) acc[c] = Bi1[c];
        for (int ky = 0; ky < 5; ++ky) {
            const int iy = oy + ky - 1;
            if (iy < 0 || iy > 6) continue;
            for (int kx = 0; kx < 5; ++kx) {
                const int ix = ox + kx - 1;
                if (ix < 0 || ix > 6) continue;
                const float* xp = &Xs[s * 147 + (iy * 7 + ix) * 3];
                const float* wp = &W1[(ky * 5 + kx) * 24];
#pragma unroll
                for (int ci = 0; ci < 3; ++ci) {
                    const float xv = xp[ci];
#pragma unroll
                    for (int co = 0; co < 8; ++co) acc[co] += xv * wp[ci * 8 + co];
                }
            }
        }
#pragma unroll
        for (int co = 0; co < 8; ++co) O1[s * 200 + p * 8 + co] = acc[co];
    }
    __syncthreads();

    // conv2: 3x3x8 -> 12ch, pad 1 (5x5 -> 5x5)
    for (int p = t; p < 25; p += 16) {
        const int oy = p / 5, ox = p % 5;
        float acc[12];
#pragma unroll
        for (int c = 0; c < 12; ++c) acc[c] = Bi2[c];
        for (int ky = 0; ky < 3; ++ky) {
            const int iy = oy + ky - 1;
            if (iy < 0 || iy > 4) continue;
            for (int kx = 0; kx < 3; ++kx) {
                const int ix = ox + kx - 1;
                if (ix < 0 || ix > 4) continue;
                const float* ip = &O1[s * 200 + (iy * 5 + ix) * 8];
                const float* wp = &W2[(ky * 3 + kx) * 96];
#pragma unroll
                for (int ci = 0; ci < 8; ++ci) {
                    const float v = ip[ci];
#pragma unroll
                    for (int co = 0; co < 12; ++co) acc[co] += v * wp[ci * 12 + co];
                }
            }
        }
#pragma unroll
        for (int co = 0; co < 12; ++co) O2[s * 300 + p * 12 + co] = acc[co];
    }
    __syncthreads();

    // attention: one query per thread (t < 4). K = [O2[:4], S], V = [O2[4:12], S]
    if (t < 4) {
        float q[8];
#pragma unroll
        for (int k = 0; k < 8; ++k) q[k] = qws[(size_t)b * 32 + t * 8 + k];
        float logit[25];
        float mx = -1e30f;
#pragma unroll
        for (int p = 0; p < 25; ++p) {
            const float* o2 = &O2[s * 300 + p * 12];
            const float* sb = &Sb[p * 4];
            const float d = q[0] * o2[0] + q[1] * o2[1] + q[2] * o2[2] + q[3] * o2[3] +
                            q[4] * sb[0] + q[5] * sb[1] + q[6] * sb[2] + q[7] * sb[3];
            logit[p] = d;
            mx = fmaxf(mx, d);
        }
        float sum = 0.f;
#pragma unroll
        for (int p = 0; p < 25; ++p) { const float e = __expf(logit[p] - mx); logit[p] = e; sum += e; }
        const float inv = 1.f / sum;
        float a[12];
#pragma unroll
        for (int d = 0; d < 12; ++d) a[d] = 0.f;
#pragma unroll
        for (int p = 0; p < 25; ++p) {
            const float w = logit[p] * inv;
            const float* o2 = &O2[s * 300 + p * 12 + 4];
            const float* sb = &Sb[p * 4];
#pragma unroll
            for (int d = 0; d < 8; ++d) a[d] += w * o2[d];
#pragma unroll
            for (int d = 0; d < 4; ++d) a[8 + d] += w * sb[d];
        }
#pragma unroll
        for (int d = 0; d < 12; ++d) Aq[s * 48 + t * 12 + d] = a[d];
    }
    __syncthreads();

    // head layer 0: 48 -> 32, relu (2 outputs / thread)
#pragma unroll
    for (int ii = 0; ii < 2; ++ii) {
        const int i = t * 2 + ii;
        float h = ab0[i];
        for (int x = 0; x < 48; ++x) h += Aq[s * 48 + x] * aw0[x * 32 + i];
        Hh[s * 32 + i] = h > 0.f ? h : 0.f;
    }
    __syncthreads();

    // head layer 1: 32 -> 64 (4 outputs / thread)
#pragma unroll
    for (int oo = 0; oo < 4; ++oo) {
        const int o = t * 4 + oo;
        float v = ab1[o];
        for (int x = 0; x < 32; ++x) v += Hh[s * 32 + x] * aw1[x * 64 + o];
        out[(size_t)b * 64 + o] = v;
    }
}

// ---------------------------------------------------------------------------
// Launch: weight prep (5 transposes) -> MLP (WMMA) -> conv/attention/head
// ---------------------------------------------------------------------------
extern "C" void kernel_launch(void* const* d_in, const int* in_sizes, int n_in,
                              void* d_out, int out_size, void* d_ws, size_t ws_size,
                              hipStream_t stream) {
    const float* X   = (const float*)d_in[0];
    const float* hid = (const float*)d_in[1];
    const float* c1w = (const float*)d_in[2];
    const float* c1b = (const float*)d_in[3];
    const float* c2w = (const float*)d_in[4];
    const float* c2b = (const float*)d_in[5];
    const float* qw[5] = {(const float*)d_in[6], (const float*)d_in[8],
                          (const float*)d_in[10], (const float*)d_in[12],
                          (const float*)d_in[14]};
    const float* qb[5] = {(const float*)d_in[7], (const float*)d_in[9],
                          (const float*)d_in[11], (const float*)d_in[13],
                          (const float*)d_in[15]};
    const float* aw0 = (const float*)d_in[16];
    const float* ab0 = (const float*)d_in[17];
    const float* aw1 = (const float*)d_in[18];
    const float* ab1 = (const float*)d_in[19];
    float* outp = (float*)d_out;

    const int B = in_sizes[0] / 147;  // X is (B,7,7,3)

    _Float16* wt  = (_Float16*)d_ws;
    float*    qws = (float*)((char*)d_ws + WT_BYTES);  // [B][32]

    // Weight prep: Wt[N][K] = qw[K][N] in f16
    const int Kd[5] = {256, 256, 128, 64, 32};
    const int Nd[5] = {256, 128, 64, 32, 32};
    const int off[5] = {OFF0, OFF1, OFF2, OFF3, OFF4};
    for (int l = 0; l < 5; ++l) {
        const int n = Kd[l] * Nd[l];
        wprep_kernel<<<(n + 255) / 256, 256, 0, stream>>>(qw[l], wt + off[l], Kd[l], Nd[l]);
    }

    // q-MLP with WMMA: 64 rows per workgroup
    mlp_kernel<<<B / 64, 256, 0, stream>>>(hid, wt, qb[0], qb[1], qb[2], qb[3], qb[4], qws);

    // conv + attention + head: 16 samples per workgroup
    conv_attn_kernel<<<B / 16, 256, 0, stream>>>(X, c1w, c1b, c2w, c2b, qws,
                                                 aw0, ab0, aw1, ab1, outp);
}
